// SelfAttention_2508260901348
// MI455X (gfx1250) — matrix operand order repaired and checked by reading, compiled, not run on hardware
//
#include <hip/hip_runtime.h>
#include <hip/hip_bf16.h>

typedef __attribute__((ext_vector_type(4)))  _Float16 v4h;
typedef __attribute__((ext_vector_type(8)))  _Float16 v8h;
typedef __attribute__((ext_vector_type(16))) _Float16 v16h;
typedef __attribute__((ext_vector_type(8)))  float    v8f;
typedef __attribute__((ext_vector_type(4)))  int      v4i;

#define B_   8
#define S_   2048
#define D_   1024
#define TILE 128      // output tile per workgroup (128x128)
#define KB   64       // K-panel depth staged in LDS
#define RS   72       // padded LDS row stride in halves (36 dwords -> conflict-free b128 reads)
#define NT_  (S_ / TILE)               // 16 tiles per dim
#define NTRI (NT_ * (NT_ + 1) / 2)     // 136 upper-triangular tiles

#if __has_builtin(__builtin_amdgcn_global_load_async_to_lds_b128)
#define HAVE_ASYNC_LDS 1
#else
#define HAVE_ASYNC_LDS 0
#endif

#if __has_builtin(__builtin_amdgcn_tensor_load_to_lds)
#define HAVE_TDM 1
#else
#define HAVE_TDM 0
#endif

typedef __attribute__((address_space(1))) v4i* glb_v4i_p;
typedef __attribute__((address_space(3))) v4i* lds_v4i_p;

__device__ __forceinline__ void wait_async0() {
#if __has_builtin(__builtin_amdgcn_s_wait_asynccnt)
    __builtin_amdgcn_s_wait_asynccnt(0);
#else
    asm volatile("s_wait_asynccnt 0" ::: "memory");
#endif
}

__device__ __forceinline__ void wait_tensor0() {
#if __has_builtin(__builtin_amdgcn_s_wait_tensorcnt)
    __builtin_amdgcn_s_wait_tensorcnt(0);
#else
    asm volatile("s_wait_tensorcnt 0x0" ::: "memory");
#endif
}

__device__ __forceinline__ unsigned lds_byte_off(void* p) {
    return (unsigned)(uintptr_t)(__attribute__((address_space(3))) void*)p;
}

#if HAVE_TDM
typedef __attribute__((ext_vector_type(4))) unsigned int u32x4;
typedef __attribute__((ext_vector_type(8))) int          i32x8;
typedef __attribute__((ext_vector_type(4))) int          i32x4;

// TDM load of a 128-row x 64-half tile (row stride D_ halves) from global f16
// into LDS with +16B padding every 128B row -> padded row stride 144B (= RS*2).
__device__ __forceinline__ void tdm_load_tile(unsigned lds_off, const _Float16* gsrc) {
    unsigned long long ga = (unsigned long long)(uintptr_t)gsrc;
    u32x4 g0;
    g0[0] = 1u;                                  // count=1 (valid descriptor)
    g0[1] = lds_off;                             // lds_addr (bytes)
    g0[2] = (unsigned)ga;                        // global_addr[31:0]
    g0[3] = (unsigned)(ga >> 32) | (2u << 30);   // global_addr[56:32] | type=2
    i32x8 g1;
    g1[0] = (int)((1u << 16)      // data_size = 2 bytes
                | (1u << 20)      // pad_enable
                | (4u << 22)      // pad_interval: 32 dwords (=128B row)
                | (3u << 25));    // pad_amount: 4 dwords (=16B)
    g1[1] = (int)((unsigned)(D_ & 0xFFFF) << 16);   // tensor_dim0 lo16 (1024)
    g1[2] = (int)((unsigned)(S_ & 0xFFFF) << 16);   // dim0 hi16=0 | tensor_dim1 lo16 (2048)
    g1[3] = (int)(64u << 16);                       // dim1 hi16=0 | tile_dim0=64
    g1[4] = 128;                                    // tile_dim1=128 | tile_dim2=0
    g1[5] = D_;                                     // tensor_dim0_stride lo32 (1024)
    g1[6] = 0;                                      // stride hi16 | dim1_stride lo16
    g1[7] = 0;                                      // dim1_stride hi32
    i32x4 zz = {0, 0, 0, 0};
#if __clang_major__ >= 23
    i32x8 zz8 = {0, 0, 0, 0, 0, 0, 0, 0};
    __builtin_amdgcn_tensor_load_to_lds(g0, g1, zz, zz, zz8, 0);
#else
    __builtin_amdgcn_tensor_load_to_lds(g0, g1, zz, zz, 0);
#endif
}
#endif  // HAVE_TDM

// ---------------------------------------------------------------------------
// Kernel 1: passthrough copy of inputs + optional f16 shadow copy into d_ws
// ---------------------------------------------------------------------------
__global__ __launch_bounds__(256) void prep_kernel(const float4* __restrict__ in,
                                                   float4* __restrict__ out,
                                                   v4h* __restrict__ xh,
                                                   int n4, int write_h) {
    int i = blockIdx.x * 256 + threadIdx.x;
    if (i < n4) {
        float4 v = in[i];
        out[i] = v;
        if (write_h) {
            v4h h = {(_Float16)v.x, (_Float16)v.y, (_Float16)v.z, (_Float16)v.w};
            xh[i] = h;
        }
    }
}

// ---------------------------------------------------------------------------
// Kernel 2: upper-triangular tiles of scores_raw = X @ X^T / S via WMMA f16.
//   grid = (136, B), block = 256 (8 wave32 waves, 2x4 wave grid).
//   Double-buffered LDS panels:
//     MODE 1: A panel via TDM tensor_load_to_lds (wave 0, TENSORcnt),
//             B panel via global_load_async_to_lds_b128 (ASYNCcnt),
//             staged from the f16 shadow in d_ws.
//     MODE 0: synchronous fp32->f16 staging straight from X.
// ---------------------------------------------------------------------------
template <int MODE>
__global__ __launch_bounds__(256) void xxt_wmma_kernel(const float* __restrict__ X,
                                                       const _Float16* __restrict__ Xh,
                                                       float* __restrict__ scores) {
    __shared__ _Float16 ldsA[2][TILE * RS];
    __shared__ _Float16 ldsB[2][TILE * RS];

    // map blockIdx.x -> upper-triangular tile (tm, tn), tn >= tm
    int ti = blockIdx.x, tm = 0;
    while (ti >= NT_ - tm) { ti -= NT_ - tm; ++tm; }
    const int tn = tm + ti;

    const int b  = blockIdx.y;
    const int s0 = tm * TILE;
    const int t0 = tn * TILE;
    const float*    Xb  = X  + (size_t)b * S_ * D_;
    const _Float16* Xhb = Xh + (size_t)b * S_ * D_;

    const int tid  = threadIdx.x;
    const int lane = tid & 31;
    const int wave = tid >> 5;
    const int wm   = wave >> 2;      // 0..1 : wave row (64 rows)
    const int wn   = wave & 3;       // 0..3 : wave col (32 cols)
    const int lrow = lane & 15;      // M / N within a 16x16 tile
    const int hlf  = lane >> 4;      // half-wave: selects K-chunk

    v8f acc[4][2];
    const v8f vzero = {0.f, 0.f, 0.f, 0.f, 0.f, 0.f, 0.f, 0.f};
#pragma unroll
    for (int mt = 0; mt < 4; ++mt)
#pragma unroll
        for (int nt = 0; nt < 2; ++nt) acc[mt][nt] = vzero;

    // ---- staging into buffer `bf` for K-offset `kb` (no wait) ----
    auto stage = [&](int bf, int kb) {
        if (MODE == 1) {
#if HAVE_TDM
            // A panel: one TDM descriptor, issued by wave 0 only
            if (tid == 0)
                tdm_load_tile(lds_byte_off(&ldsA[bf][0]),
                              Xhb + (size_t)s0 * D_ + kb);
#elif HAVE_ASYNC_LDS
#pragma unroll
            for (int i = 0; i < 4; ++i) {
                const int c = tid + i * 256, row = c >> 3, ch = c & 7;
                __builtin_amdgcn_global_load_async_to_lds_b128(
                    (glb_v4i_p)(void*)(_Float16*)(Xhb + (size_t)(s0 + row) * D_ + kb + ch * 8),
                    (lds_v4i_p)(void*)&ldsA[bf][row * RS + ch * 8], 0, 0);
            }
#endif
#if HAVE_ASYNC_LDS
            // B panel: per-lane async copies (1024 x 16B chunks, 4/thread)
#pragma unroll
            for (int i = 0; i < 4; ++i) {
                const int c = tid + i * 256, row = c >> 3, ch = c & 7;
                __builtin_amdgcn_global_load_async_to_lds_b128(
                    (glb_v4i_p)(void*)(_Float16*)(Xhb + (size_t)(t0 + row) * D_ + kb + ch * 8),
                    (lds_v4i_p)(void*)&ldsB[bf][row * RS + ch * 8], 0, 0);
            }
#elif HAVE_TDM
            if (tid == 0)
                tdm_load_tile(lds_byte_off(&ldsB[bf][0]),
                              Xhb + (size_t)t0 * D_ + kb);
#endif
        } else {
            // synchronous staging: fp32 global -> cvt f16 -> LDS
#pragma unroll
            for (int i = 0; i < 8; ++i) {
                const int c = tid + i * 256, row = c >> 4, c4 = c & 15;
                const float4 ga = *(const float4*)(Xb + (size_t)(s0 + row) * D_ + kb + c4 * 4);
                const float4 gb = *(const float4*)(Xb + (size_t)(t0 + row) * D_ + kb + c4 * 4);
                v4h ha = {(_Float16)ga.x, (_Float16)ga.y, (_Float16)ga.z, (_Float16)ga.w};
                v4h hb = {(_Float16)gb.x, (_Float16)gb.y, (_Float16)gb.z, (_Float16)gb.w};
                *(v4h*)&ldsA[bf][row * RS + c4 * 4] = ha;
                *(v4h*)&ldsB[bf][row * RS + c4 * 4] = hb;
            }
        }
    };
    auto wait_stage = [&]() {
        if (MODE == 1) {
            wait_async0();
            wait_tensor0();
        }
    };

    // prologue: stage first panel
    stage(0, 0);
    wait_stage();
    __syncthreads();

    int buf = 0;
    for (int kb = 0; kb < D_; kb += KB) {
        // overlap: issue DMA for next K-block into the other buffer
        if (kb + KB < D_) stage(buf ^ 1, kb + KB);

        // compute on current buffer: load all fragments, then 16 WMMAs
        const _Float16* LA = &ldsA[buf][0];
        const _Float16* LB = &ldsB[buf][0];
        union Frag { v16h v; v8h h[2]; };
        Frag af[2][4], bfr[2][2];
#pragma unroll
        for (int kk = 0; kk < 2; ++kk) {
            const int ks = kk * 32;
#pragma unroll
            for (int mt = 0; mt < 4; ++mt) {
                const _Float16* p = LA + (wm * 64 + mt * 16 + lrow) * RS + ks + hlf * 8;
                af[kk][mt].h[0] = *(const v8h*)p;
                af[kk][mt].h[1] = *(const v8h*)(p + 16);
            }
#pragma unroll
            for (int nt = 0; nt < 2; ++nt) {
                const _Float16* p = LB + (wn * 32 + nt * 16 + lrow) * RS + ks + hlf * 8;
                bfr[kk][nt].h[0] = *(const v8h*)p;
                bfr[kk][nt].h[1] = *(const v8h*)(p + 16);
            }
        }
#pragma unroll
        for (int kk = 0; kk < 2; ++kk)
#pragma unroll
            for (int mt = 0; mt < 4; ++mt)
#pragma unroll
                for (int nt = 0; nt < 2; ++nt)
                    acc[mt][nt] = __builtin_amdgcn_wmma_f32_16x16x32_f16(
                        false, af[kk][mt].v, false, bfr[kk][nt].v,
                        (short)0, acc[mt][nt], false, false);

        wait_stage();      // next panel landed in LDS
        __syncthreads();   // all waves done reading `buf`, writes visible
        buf ^= 1;
    }

    // Epilogue: scale by 1/S; write tile, and mirrored tile when off-diagonal.
    // C/D layout: VGPR r -> M = r + 8*hlf, lane -> N = lrow.
    const float scale = 1.0f / (float)S_;
    const bool  diag  = (tm == tn);
    float* Sb = scores + (size_t)b * S_ * S_;
#pragma unroll
    for (int mt = 0; mt < 4; ++mt) {
        const int rbase = s0 + wm * 64 + mt * 16 + hlf * 8;   // global s index base
#pragma unroll
        for (int nt = 0; nt < 2; ++nt) {
            const int col = t0 + wn * 32 + nt * 16 + lrow;    // global t index
            float v[8];
#pragma unroll
            for (int r = 0; r < 8; ++r) v[r] = acc[mt][nt][r] * scale;
#pragma unroll
            for (int r = 0; r < 8; ++r)
                Sb[(size_t)(rbase + r) * S_ + col] = v[r];
            if (!diag) {
                float4 lo = {v[0], v[1], v[2], v[3]};
                float4 hi = {v[4], v[5], v[6], v[7]};
                float* mp = &Sb[(size_t)col * S_ + rbase];
                *(float4*)(mp)     = lo;
                *(float4*)(mp + 4) = hi;
            }
        }
    }
}

// ---------------------------------------------------------------------------
// Kernel 3: in-place row softmax over scores (B*S rows of length S)
// ---------------------------------------------------------------------------
__global__ __launch_bounds__(256) void softmax_rows_kernel(float* __restrict__ scores) {
    __shared__ float red[256];
    float* p = scores + (size_t)blockIdx.x * S_;
    const int t = threadIdx.x;

    float vals[8];
    float m = -3.402823466e+38f;
#pragma unroll
    for (int i = 0; i < 8; ++i) {
        vals[i] = p[t + i * 256];
        m = fmaxf(m, vals[i]);
    }
    red[t] = m;
    __syncthreads();
    for (int off = 128; off > 0; off >>= 1) {
        if (t < off) red[t] = fmaxf(red[t], red[t + off]);
        __syncthreads();
    }
    m = red[0];
    __syncthreads();

    float s = 0.f;
#pragma unroll
    for (int i = 0; i < 8; ++i) {
        vals[i] = __expf(vals[i] - m);
        s += vals[i];
    }
    red[t] = s;
    __syncthreads();
    for (int off = 128; off > 0; off >>= 1) {
        if (t < off) red[t] += red[t + off];
        __syncthreads();
    }
    const float inv = 1.0f / red[0];
#pragma unroll
    for (int i = 0; i < 8; ++i) p[t + i * 256] = vals[i] * inv;
}

// ---------------------------------------------------------------------------
extern "C" void kernel_launch(void* const* d_in, const int* in_sizes, int n_in,
                              void* d_out, int out_size, void* d_ws, size_t ws_size,
                              hipStream_t stream) {
    (void)in_sizes; (void)n_in; (void)out_size;
    const float* X  = (const float*)d_in[0];
    float* out      = (float*)d_out;
    float* scores   = out + (size_t)B_ * S_ * D_;   // second tuple element
    _Float16* Xh    = (_Float16*)d_ws;

    const size_t need_ws = (size_t)B_ * S_ * D_ * sizeof(_Float16);
    const bool dma_ok = (HAVE_ASYNC_LDS || HAVE_TDM);
    const bool use_ws = dma_ok && (d_ws != nullptr) && (ws_size >= need_ws);

    // 1) inputs passthrough (+ f16 shadow for the DMA staging path)
    const int n4 = (B_ * S_ * D_) / 4;
    prep_kernel<<<(n4 + 255) / 256, 256, 0, stream>>>(
        (const float4*)X, (float4*)out, (v4h*)Xh, n4, use_ws ? 1 : 0);

    // 2) raw scores via WMMA (upper triangle + mirror)
    dim3 grid(NTRI, B_, 1);
    if (use_ws)
        xxt_wmma_kernel<1><<<grid, 256, 0, stream>>>(X, Xh, scores);
    else
        xxt_wmma_kernel<0><<<grid, 256, 0, stream>>>(X, Xh, scores);

    // 3) row-wise softmax in place
    softmax_rows_kernel<<<B_ * S_, 256, 0, stream>>>(scores);
}
